// VAE_GNN_32667521253736
// MI455X (gfx1250) — compile-verified, matrix operand order checked by reading, not run on hardware
//
#include <hip/hip_runtime.h>
#include <stdint.h>

// ---------------- problem constants ----------------
static constexpr int kN   = 2048;    // nodes
static constexpr int kE   = 65536;   // edges
static constexpr int HID  = 256;
static constexpr int ZD   = 64;
static constexpr int OUTD = 24;
static constexpr int ENC  = 536;     // 2*HID + OUTD
static constexpr int DEC  = 576;     // ZD + 2*HID

typedef __attribute__((ext_vector_type(16))) _Float16 v16h;
typedef __attribute__((ext_vector_type(8)))  _Float16 h8;    // one 16B load
typedef __attribute__((ext_vector_type(8)))  float    v8f;

static inline int rup(int x, int m) { return (x + m - 1) / m * m; }

// =====================================================================
// Operand packing:
//   A16[r*K32 + k] = (k<K)   ? f16(A[r*lda + k]) : 0        (row-major, padded K)
//   WT [n*K32 + k] = (n<Nn && k<K) ? f16(W[k*Nn + n]) : 0   (transposed, padded)
// Fragments become contiguous 8-half vectors -> pure b128 loads, no guards.
// =====================================================================
__global__ void k_prep_a(const float* __restrict__ A, int lda, int K, int K32,
                         int M, _Float16* __restrict__ A16)
{
    int i = blockIdx.x * blockDim.x + threadIdx.x;
    if (i >= M * K32) return;
    int r = i / K32, k = i % K32;
    A16[i] = (k < K) ? (_Float16)A[(size_t)r * lda + k] : (_Float16)0.0f;
}

__global__ void k_prep_w(const float* __restrict__ W, int K, int Nn,
                         int K32, int N32, _Float16* __restrict__ WT)
{
    int i = blockIdx.x * blockDim.x + threadIdx.x;
    if (i >= N32 * K32) return;
    int n = i / K32, k = i % K32;
    WT[i] = (n < Nn && k < K) ? (_Float16)W[(size_t)k * Nn + n] : (_Float16)0.0f;
}

// =====================================================================
// WMMA GEMM: C[M,Nn] = act( A @ W + bias ), f16 in / f32 acc.
// One wave per 32x32 output tile: 4 accumulators, 4 WMMAs per 32-K chunk,
// guard-free b128 fragment loads from packed A16 / WT.
// act: 0 none, 1 relu, 2 leaky_relu(0.01)
// =====================================================================
__device__ __forceinline__ v16h mkfrag(h8 lo, h8 hi) {
    v16h r;
#pragma unroll
    for (int j = 0; j < 8; ++j) { r[j] = lo[j]; r[8 + j] = hi[j]; }
    return r;
}

__device__ __forceinline__ void store_tile(float* __restrict__ C, int ldc, int Nn,
                                           const float* __restrict__ bias, int act,
                                           int mbase, int nbase, int half, int l15,
                                           v8f c)
{
    const int col = nbase + l15;
    if (col >= Nn) return;
    const float bv = bias ? bias[col] : 0.0f;
    const int rowbase = mbase + half * 8;
#pragma unroll
    for (int r = 0; r < 8; ++r) {
        float v = c[r] + bv;
        if (act == 1)      v = v > 0.f ? v : 0.f;
        else if (act == 2) v = v > 0.f ? v : 0.01f * v;
        C[(size_t)(rowbase + r) * ldc + col] = v;
    }
}

__global__ void k_gemm_wmma(const _Float16* __restrict__ A16,
                            const _Float16* __restrict__ WT,
                            const float* __restrict__ bias,
                            float* __restrict__ C, int ldc,
                            int K32, int Nn, int act)
{
    const int lane = threadIdx.x & 31;
    const int m0 = blockIdx.x * 32;
    const int n0 = blockIdx.y * 32;
    const int half = lane >> 4;      // 0: lanes 0-15, 1: lanes 16-31
    const int l15  = lane & 15;

    const h8* __restrict__ pa0 = (const h8*)(A16 + (size_t)(m0 + l15)      * K32);
    const h8* __restrict__ pa1 = (const h8*)(A16 + (size_t)(m0 + 16 + l15) * K32);
    const h8* __restrict__ pb0 = (const h8*)(WT  + (size_t)(n0 + l15)      * K32);
    const h8* __restrict__ pb1 = (const h8*)(WT  + (size_t)(n0 + 16 + l15) * K32);

    v8f c00 = {}, c01 = {}, c10 = {}, c11 = {};

    const int chunks = K32 >> 5;     // 32 halves (4 h8 blocks) per chunk
    for (int kc = 0; kc < chunks; ++kc) {
        const int base = kc * 4 + half;      // lo block; hi block = base + 2
        // prefetch next chunk (global_prefetch_b8)
        __builtin_prefetch((const void*)(pa0 + base + 4), 0, 1);
        __builtin_prefetch((const void*)(pb0 + base + 4), 0, 1);

        v16h a0 = mkfrag(pa0[base], pa0[base + 2]);
        v16h a1 = mkfrag(pa1[base], pa1[base + 2]);
        v16h b0 = mkfrag(pb0[base], pb0[base + 2]);
        v16h b1 = mkfrag(pb1[base], pb1[base + 2]);

        c00 = __builtin_amdgcn_wmma_f32_16x16x32_f16(false, a0, false, b0, (short)0, c00, false, false);
        c01 = __builtin_amdgcn_wmma_f32_16x16x32_f16(false, a0, false, b1, (short)0, c01, false, false);
        c10 = __builtin_amdgcn_wmma_f32_16x16x32_f16(false, a1, false, b0, (short)0, c10, false, false);
        c11 = __builtin_amdgcn_wmma_f32_16x16x32_f16(false, a1, false, b1, (short)0, c11, false, false);
    }

    store_tile(C, ldc, Nn, bias, act, m0,      n0,      half, l15, c00);
    store_tile(C, ldc, Nn, bias, act, m0,      n0 + 16, half, l15, c01);
    store_tile(C, ldc, Nn, bias, act, m0 + 16, n0,      half, l15, c10);
    store_tile(C, ldc, Nn, bias, act, m0 + 16, n0 + 16, half, l15, c11);
}

// =====================================================================
// Fused conv stack per image, all intermediates in LDS (320KB/WGP on CDNA5).
// 3x112x112 ->(5,s2) 10x54x54 ->(5,s2) 20x25x25 ->(5,s1) 10x21x21
// ->(3,s1) 1x19x19 -> flatten 361 -> fc 256  (relu after each conv, fc linear)
// =====================================================================
static constexpr int C1O = 10 * 54 * 54;   // 29160
static constexpr int C2O = 20 * 25 * 25;   // 12500
static constexpr int C3O = 10 * 21 * 21;   // 4410
static constexpr int C4O = 19 * 19;        // 361
static constexpr int CONV_LDS_BYTES = (C1O + C2O + C3O + C4O) * 4; // 185724

__global__ void k_conv_stack(const float* __restrict__ maps,
                             const float* __restrict__ K1, const float* __restrict__ B1,
                             const float* __restrict__ K2, const float* __restrict__ B2,
                             const float* __restrict__ K3, const float* __restrict__ B3,
                             const float* __restrict__ K4, const float* __restrict__ B4,
                             const float* __restrict__ fcW, const float* __restrict__ fcB,
                             float* __restrict__ memb)
{
    extern __shared__ float sm[];
    float* b1 = sm;
    float* b2 = b1 + C1O;
    float* b3 = b2 + C2O;
    float* b4 = b3 + C3O;

    const int n = blockIdx.x;
    const int t = threadIdx.x;
    const float* in = maps + (size_t)n * 3 * 112 * 112;

    for (int idx = t; idx < C1O; idx += 256) {
        int c = idx / (54 * 54), rem = idx % (54 * 54);
        int y = rem / 54, x = rem % 54;
        float acc = B1[c];
        for (int ci = 0; ci < 3; ++ci)
            for (int ky = 0; ky < 5; ++ky) {
                const float* ip = in + ci * 112 * 112 + (2 * y + ky) * 112 + 2 * x;
                const float* kp = K1 + ((c * 3 + ci) * 5 + ky) * 5;
#pragma unroll
                for (int kx = 0; kx < 5; ++kx) acc += ip[kx] * kp[kx];
            }
        b1[idx] = acc > 0.f ? acc : 0.f;
    }
    __syncthreads();

    for (int idx = t; idx < C2O; idx += 256) {
        int c = idx / (25 * 25), rem = idx % (25 * 25);
        int y = rem / 25, x = rem % 25;
        float acc = B2[c];
        for (int ci = 0; ci < 10; ++ci)
            for (int ky = 0; ky < 5; ++ky) {
                const float* ip = b1 + ci * 54 * 54 + (2 * y + ky) * 54 + 2 * x;
                const float* kp = K2 + ((c * 10 + ci) * 5 + ky) * 5;
#pragma unroll
                for (int kx = 0; kx < 5; ++kx) acc += ip[kx] * kp[kx];
            }
        b2[idx] = acc > 0.f ? acc : 0.f;
    }
    __syncthreads();

    for (int idx = t; idx < C3O; idx += 256) {
        int c = idx / (21 * 21), rem = idx % (21 * 21);
        int y = rem / 21, x = rem % 21;
        float acc = B3[c];
        for (int ci = 0; ci < 20; ++ci)
            for (int ky = 0; ky < 5; ++ky) {
                const float* ip = b2 + ci * 25 * 25 + (y + ky) * 25 + x;
                const float* kp = K3 + ((c * 20 + ci) * 5 + ky) * 5;
#pragma unroll
                for (int kx = 0; kx < 5; ++kx) acc += ip[kx] * kp[kx];
            }
        b3[idx] = acc > 0.f ? acc : 0.f;
    }
    __syncthreads();

    for (int idx = t; idx < C4O; idx += 256) {
        int y = idx / 19, x = idx % 19;
        float acc = B4[0];
        for (int ci = 0; ci < 10; ++ci)
            for (int ky = 0; ky < 3; ++ky) {
                const float* ip = b3 + ci * 21 * 21 + (y + ky) * 21 + x;
                const float* kp = K4 + (ci * 3 + ky) * 3;
#pragma unroll
                for (int kx = 0; kx < 3; ++kx) acc += ip[kx] * kp[kx];
            }
        b4[idx] = acc > 0.f ? acc : 0.f;
    }
    __syncthreads();

    {
        int o = t;
        float acc = fcB[o];
        for (int j = 0; j < C4O; ++j) acc += b4[j] * fcW[j * 256 + o];
        memb[(size_t)n * 256 + o] = acc;
    }
}

// =====================================================================
// GAT helpers
// =====================================================================
__global__ void k_node_dots(const float* __restrict__ z, int ldz, int d,
                            const float* __restrict__ a,
                            float* __restrict__ za, float* __restrict__ zd)
{
    const int n = blockIdx.x, lane = threadIdx.x;
    float s1 = 0.f, s2 = 0.f;
    for (int j = lane; j < d; j += 32) {
        float v = z[(size_t)n * ldz + j];
        s1 += v * a[j];
        s2 += v * a[d + j];
    }
    for (int off = 16; off > 0; off >>= 1) {
        s1 += __shfl_down(s1, off, 32);
        s2 += __shfl_down(s2, off, 32);
    }
    if (lane == 0) { za[n] = s1; zd[n] = s2; }
}

// s[0] = dot(b_emb, a3), s[1] = dot(W_emb[0,:], a3)  (edge-feature collapse)
__global__ void k_edge_scalar(const float* __restrict__ Wemb,
                              const float* __restrict__ bemb,
                              const float* __restrict__ a3, int d,
                              float* __restrict__ s)
{
    __shared__ float r0[256], r1[256];
    const int t = threadIdx.x;
    float p0 = 0.f, p1 = 0.f;
    for (int j = t; j < d; j += 256) {
        float av = a3[j];
        p0 += bemb[j] * av;
        p1 += Wemb[j] * av;
    }
    r0[t] = p0; r1[t] = p1; __syncthreads();
    for (int o = 128; o > 0; o >>= 1) {
        if (t < o) { r0[t] += r0[t + o]; r1[t] += r1[t + o]; }
        __syncthreads();
    }
    if (t == 0) { s[0] = r0[0]; s[1] = r1[0]; }
}

__device__ __forceinline__ unsigned f_enc(float f) {
    unsigned u = __float_as_uint(f);
    return (u & 0x80000000u) ? ~u : (u | 0x80000000u);
}
__device__ __forceinline__ float f_dec(unsigned u) {
    return (u & 0x80000000u) ? __uint_as_float(u & 0x7FFFFFFFu)
                             : __uint_as_float(~u);
}

__global__ void k_edge_logits(const int* __restrict__ src, const int* __restrict__ dst,
                              const float* __restrict__ za, const float* __restrict__ zd,
                              const float* __restrict__ ew, const float* __restrict__ s,
                              int use_ew, float* __restrict__ logit,
                              unsigned* __restrict__ mU, int E)
{
    const int e = blockIdx.x * blockDim.x + threadIdx.x;
    if (e >= E) return;
    float l = za[src[e]] + zd[dst[e]];
    if (use_ew) l += ew[e] * s[1] + s[0];
    l = l > 0.f ? l : 0.01f * l;          // leaky_relu(0.01)
    logit[e] = l;
    atomicMax(&mU[dst[e]], f_enc(l));
}

__global__ void k_edge_exp(const float* __restrict__ logit, const int* __restrict__ dst,
                           const unsigned* __restrict__ mU,
                           float* __restrict__ ex, float* __restrict__ ssum, int E)
{
    const int e = blockIdx.x * blockDim.x + threadIdx.x;
    if (e >= E) return;
    float m = f_dec(mU[dst[e]]);
    float v = expf(logit[e] - m);
    ex[e] = v;
    atomicAdd(&ssum[dst[e]], v);
}

__global__ void k_edge_agg(const float* __restrict__ ex, const float* __restrict__ ssum,
                           const int* __restrict__ src, const int* __restrict__ dst,
                           const float* __restrict__ z, int ldz, int d,
                           float* __restrict__ agg, int ldagg)
{
    const int e = blockIdx.x, lane = threadIdx.x;
    const int sn = src[e], dn = dst[e];
    const float w = ex[e] / ssum[dn];
    for (int j = lane; j < d; j += 32)
        atomicAdd(&agg[(size_t)dn * ldagg + j], w * z[(size_t)sn * ldz + j]);
}

// h = relu(h + (hs + agg) * snorm_n)   (in place)
__global__ void k_gat_finish(float* __restrict__ h, int ldh,
                             const float* __restrict__ hs, const float* __restrict__ agg,
                             int ldz, const float* __restrict__ snorm,
                             int d, int n_nodes)
{
    const int i = blockIdx.x * blockDim.x + threadIdx.x;
    if (i >= n_nodes * d) return;
    const int n = i / d, j = i % d;
    float out = (hs[(size_t)n * ldz + j] + agg[(size_t)n * ldz + j]) * snorm[n];
    float v = h[(size_t)n * ldh + j] + out;
    h[(size_t)n * ldh + j] = v > 0.f ? v : 0.f;
}

// =====================================================================
// small utilities
// =====================================================================
__global__ void k_fill(float* __restrict__ p, int n, float v) {
    int i = blockIdx.x * blockDim.x + threadIdx.x;
    if (i < n) p[i] = v;
}

__global__ void k_copy_cols(float* __restrict__ dstp, int ldd, int off,
                            const float* __restrict__ srcp, int ldsrc,
                            int w, int rows)
{
    int i = blockIdx.x * blockDim.x + threadIdx.x;
    if (i >= rows * w) return;
    int n = i / w, j = i % w;
    dstp[(size_t)n * ldd + off + j] = srcp[(size_t)n * ldsrc + j];
}

__global__ void k_reparam(const float* __restrict__ mu, const float* __restrict__ lv,
                          const float* __restrict__ eps, float* __restrict__ z, int n)
{
    int i = blockIdx.x * blockDim.x + threadIdx.x;
    if (i >= n) return;
    z[i] = mu[i] + expf(0.5f * lv[i]) * eps[i];
}

// =====================================================================
// kernel_launch
// =====================================================================
extern "C" void kernel_launch(void* const* d_in, const int* in_sizes, int n_in,
                              void* d_out, int out_size, void* d_ws, size_t ws_size,
                              hipStream_t stream)
{
    (void)in_sizes; (void)n_in; (void)out_size; (void)ws_size;

    // ---- inputs (setup_inputs dict insertion order, params flattened in order) ----
    const float* feats   = (const float*)d_in[0];
    const float* e_w     = (const float*)d_in[1];
    const float* snorm_n = (const float*)d_in[2];
    const float* gt      = (const float*)d_in[4];
    const float* maps    = (const float*)d_in[5];
    const int*   src     = (const int*)d_in[6];
    const int*   dst     = (const int*)d_in[7];
    const float* eps     = (const float*)d_in[8];
    const float* embh_W  = (const float*)d_in[9];
    const float* embh_b  = (const float*)d_in[10];
    const float* embe_W  = (const float*)d_in[11];
    const float* embe_b  = (const float*)d_in[12];
    const float* cK1 = (const float*)d_in[15];  const float* cB1 = (const float*)d_in[16];
    const float* cK2 = (const float*)d_in[17];  const float* cB2 = (const float*)d_in[18];
    const float* cK3 = (const float*)d_in[19];  const float* cB3 = (const float*)d_in[20];
    const float* cK4 = (const float*)d_in[21];  const float* cB4 = (const float*)d_in[22];
    const float* fcW = (const float*)d_in[23];  const float* fcB = (const float*)d_in[24];
    const float* ge1_Ws = (const float*)d_in[25];
    const float* ge1_Wf = (const float*)d_in[26];
    const float* ge1_a  = (const float*)d_in[27];
    const float* ge2_Ws = (const float*)d_in[28];
    const float* ge2_Wf = (const float*)d_in[29];
    const float* ge2_a  = (const float*)d_in[30];
    const float* gd1_Ws = (const float*)d_in[31];
    const float* gd1_Wf = (const float*)d_in[32];
    const float* gd1_a  = (const float*)d_in[33];
    const float* gd2_Ws = (const float*)d_in[34];
    const float* gd2_Wf = (const float*)d_in[35];
    const float* gd2_a  = (const float*)d_in[36];
    const float* encl_W = (const float*)d_in[37]; const float* encl_b = (const float*)d_in[38];
    const float* mu_W   = (const float*)d_in[39]; const float* mu_b   = (const float*)d_in[40];
    const float* lv_W   = (const float*)d_in[41]; const float* lv_b   = (const float*)d_in[42];
    const float* d1_W   = (const float*)d_in[43]; const float* d1_b   = (const float*)d_in[44];
    const float* d2_W   = (const float*)d_in[45]; const float* d2_b   = (const float*)d_in[46];
    const float* d3_W   = (const float*)d_in[47]; const float* d3_b   = (const float*)d_in[48];

    // ---- outputs ----
    float* out_recon = (float*)d_out;                         // (kN, 24)
    float* out_mu    = out_recon + (size_t)kN * OUTD;         // (kN, 64)
    float* out_lv    = out_mu    + (size_t)kN * ZD;           // (kN, 64)

    // ---- workspace layout: f32 region ----
    float* w = (float*)d_ws;
    float* maps_emb = w;                    w += (size_t)kN * HID;
    float* h_emb    = w;                    w += (size_t)kN * HID;
    float* h_enc    = w;                    w += (size_t)kN * 560;   // [h(536) | gt(24)]
    float* h_dec    = w;                    w += (size_t)kN * 640;   // [h(576) | z(64)]
    float* zb       = w;                    w += (size_t)kN * 576;
    float* hsb      = w;                    w += (size_t)kN * 576;
    float* aggb     = w;                    w += (size_t)kN * 576;
    float* zlat     = w;                    w += (size_t)kN * ZD;
    float* he       = w;                    w += (size_t)kN * 280;
    float* za       = w;                    w += kN;
    float* zd       = w;                    w += kN;
    float* mbuf     = w;                    w += kN;
    float* ssum     = w;                    w += kN;
    float* logit    = w;                    w += kE;
    float* alpha    = w;                    w += kE;
    float* svals    = w;                    w += 2;

    // ---- workspace layout: f16 region (16B aligned) ----
    _Float16* hb = (_Float16*)(((uintptr_t)w + 15) & ~(uintptr_t)15);
    _Float16* a16 = hb;                     hb += (size_t)kN * 640;  // shared packed-A buffer

    // pack a weight (transposed, K and N padded to 32)
    auto prepw = [&](const float* W, int K, int Nn) -> _Float16* {
        int K32 = rup(K, 32), N32 = rup(Nn, 32);
        _Float16* wt = hb; hb += (size_t)N32 * K32;
        int tot = N32 * K32;
        k_prep_w<<<(tot + 255) / 256, 256, 0, stream>>>(W, K, Nn, K32, N32, wt);
        return wt;
    };

    _Float16* wt_embh = prepw(embh_W, 24, HID);
    _Float16* wt_e1f  = prepw(ge1_Wf, ENC, ENC);
    _Float16* wt_e1s  = prepw(ge1_Ws, ENC, ENC);
    _Float16* wt_e2f  = prepw(ge2_Wf, ENC, ENC);
    _Float16* wt_e2s  = prepw(ge2_Ws, ENC, ENC);
    _Float16* wt_d1f  = prepw(gd1_Wf, DEC, DEC);
    _Float16* wt_d1s  = prepw(gd1_Ws, DEC, DEC);
    _Float16* wt_d2f  = prepw(gd2_Wf, DEC, DEC);
    _Float16* wt_d2s  = prepw(gd2_Ws, DEC, DEC);
    _Float16* wt_encl = prepw(encl_W, 560, 280);
    _Float16* wt_mu   = prepw(mu_W, 280, ZD);
    _Float16* wt_lv   = prepw(lv_W, 280, ZD);
    _Float16* wt_dec1 = prepw(d1_W, 640, DEC);
    _Float16* wt_dec2 = prepw(d2_W, DEC, 288);
    _Float16* wt_dec3 = prepw(d3_W, 288, OUTD);

    // pack activations (K padded to 32) into shared a16 buffer
    auto prepa = [&](const float* A, int lda, int K) {
        int K32 = rup(K, 32);
        int tot = kN * K32;
        k_prep_a<<<(tot + 255) / 256, 256, 0, stream>>>(A, lda, K, K32, kN, a16);
    };

    // GEMM on the currently packed A
    auto rung = [&](int K, _Float16* wt, const float* bias,
                    float* C, int ldc, int Nn, int act) {
        int K32 = rup(K, 32);
        dim3 g(kN / 32, (Nn + 31) / 32);
        k_gemm_wmma<<<g, dim3(32), 0, stream>>>(a16, wt, bias, C, ldc, K32, Nn, act);
    };

    // full GAT layer (h updated in place)
    auto run_gat = [&](float* h, int ldh, int d,
                       _Float16* wtF, _Float16* wtS, const float* a, int use_ew) {
        const int ldz = 576;
        prepa(h, ldh, d);
        rung(d, wtF, nullptr, zb,  ldz, d, 0);   // z  = h @ Wf
        rung(d, wtS, nullptr, hsb, ldz, d, 0);   // hs = h @ Ws
        k_node_dots<<<kN, 32, 0, stream>>>(zb, ldz, d, a, za, zd);
        if (use_ew)
            k_edge_scalar<<<1, 256, 0, stream>>>(embe_W, embe_b, a + 2 * d, d, svals);
        k_fill<<<(kN + 255) / 256, 256, 0, stream>>>(mbuf, kN, 0.f); // bits==0 == -inf encode
        k_fill<<<(kN + 255) / 256, 256, 0, stream>>>(ssum, kN, 0.f);
        k_fill<<<(kN * ldz + 255) / 256, 256, 0, stream>>>(aggb, kN * ldz, 0.f);
        k_edge_logits<<<kE / 256, 256, 0, stream>>>(src, dst, za, zd, e_w, svals,
                                                    use_ew, logit, (unsigned*)mbuf, kE);
        k_edge_exp<<<kE / 256, 256, 0, stream>>>(logit, dst, (const unsigned*)mbuf,
                                                 alpha, ssum, kE);
        k_edge_agg<<<kE, 32, 0, stream>>>(alpha, ssum, src, dst, zb, ldz, d, aggb, ldz);
        k_gat_finish<<<(kN * d + 255) / 256, 256, 0, stream>>>(h, ldh, hsb, aggb, ldz,
                                                               snorm_n, d, kN);
    };

    // 1) map encoder (fused conv pyramid in LDS)
    k_conv_stack<<<kN, 256, CONV_LDS_BYTES, stream>>>(
        maps, cK1, cB1, cK2, cB2, cK3, cB3, cK4, cB4, fcW, fcB, maps_emb);

    // 2) node embedding:  h_emb = feats @ W + b
    prepa(feats, 24, 24);
    rung(24, wt_embh, embh_b, h_emb, HID, HID, 0);

    // 3) encoder h = [maps_emb | h_emb | gt]
    k_copy_cols<<<(kN * HID + 255) / 256, 256, 0, stream>>>(h_enc, 560, 0,   maps_emb, HID, HID, kN);
    k_copy_cols<<<(kN * HID + 255) / 256, 256, 0, stream>>>(h_enc, 560, 256, h_emb,    HID, HID, kN);
    k_copy_cols<<<(kN * OUTD + 255) / 256, 256, 0, stream>>>(h_enc, 560, 512, gt,      OUTD, OUTD, kN);

    // 4) encoder GAT layers (att_ew = true)
    run_gat(h_enc, 560, ENC, wt_e1f, wt_e1s, ge1_a, 1);
    run_gat(h_enc, 560, ENC, wt_e2f, wt_e2s, ge2_a, 1);

    // 5) append gt -> enc_lin -> mu / log_var (directly into d_out)
    k_copy_cols<<<(kN * OUTD + 255) / 256, 256, 0, stream>>>(h_enc, 560, 536, gt, OUTD, OUTD, kN);
    prepa(h_enc, 560, 560);
    rung(560, wt_encl, encl_b, he, 280, 280, 2);           // leaky 0.01
    prepa(he, 280, 280);
    rung(280, wt_mu, mu_b, out_mu, ZD, ZD, 0);
    rung(280, wt_lv, lv_b, out_lv, ZD, ZD, 0);

    // 6) reparameterization: z = mu + exp(0.5*lv)*eps
    k_reparam<<<(kN * ZD + 255) / 256, 256, 0, stream>>>(out_mu, out_lv, eps, zlat, kN * ZD);

    // 7) decoder h = [maps_emb | h_emb | z]
    k_copy_cols<<<(kN * HID + 255) / 256, 256, 0, stream>>>(h_dec, 640, 0,   maps_emb, HID, HID, kN);
    k_copy_cols<<<(kN * HID + 255) / 256, 256, 0, stream>>>(h_dec, 640, 256, h_emb,    HID, HID, kN);
    k_copy_cols<<<(kN * ZD + 255) / 256, 256, 0, stream>>>(h_dec, 640, 512, zlat, ZD, ZD, kN);

    // 8) decoder GAT layers (att_ew = false)
    run_gat(h_dec, 640, DEC, wt_d1f, wt_d1s, gd1_a, 0);
    run_gat(h_dec, 640, DEC, wt_d2f, wt_d2s, gd2_a, 0);

    // 9) append z -> dec MLP -> recon_y (into d_out)
    k_copy_cols<<<(kN * ZD + 255) / 256, 256, 0, stream>>>(h_dec, 640, 576, zlat, ZD, ZD, kN);
    prepa(h_dec, 640, 640);
    rung(640, wt_dec1, d1_b, hsb, 576, DEC, 1);            // relu
    prepa(hsb, 576, 576);
    rung(576, wt_dec2, d2_b, aggb, 288, 288, 1);           // relu
    prepa(aggb, 288, 288);
    rung(288, wt_dec3, d3_b, out_recon, OUTD, OUTD, 0);
}
